// EfficientAttention_74148315398286
// MI455X (gfx1250) — matrix operand order repaired and checked by reading, not compiled, not run on hardware
//
#include <hip/hip_runtime.h>

// ---------------------------------------------------------------------------
// EfficientAttention (windowed MQA) for MI455X / gfx1250, wave32 + WMMA.
//   q  = x @ q_w^T              (16384 x 1024, K=1024) f16 GEMM
//   kv = x @ kv_w^T             (16384 x 128,  K=1024) f16 GEMM (K|V packed)
//   per (b,win,h): softmax(Q Kt * 1/8) V                WMMA attention
//   out = att @ proj_w^T + b    (16384 x 1024, K=1024) f32-out GEMM
// GEMMs: 64x64 C tile per wave (4x4 wmma register blocking, 4x A/B fragment
// reuse); K hardcoded to 1024 so fragment row strides fold into the 24-bit
// immediate offsets of global_load_b128.
// ---------------------------------------------------------------------------

typedef _Float16 half_t;
typedef __attribute__((ext_vector_type(16))) _Float16 v16h;
typedef __attribute__((ext_vector_type(8)))  _Float16 v8h;
typedef __attribute__((ext_vector_type(8)))  float    v8f;

#define DIMD    1024
#define NHEADS  16
#define HD      64
#define WIN     256
#define BATCH   4
#define SEQ     4096
#define NROWS   (BATCH * SEQ)       // 16384
#define NWIN    (SEQ / WIN)         // 16
#define QKSCALE 0.125f              // 64^-0.5
#define GEMM_K  1024                // shared K dim of all three projections

__device__ __forceinline__ v8f wmma16x16x32(v16h a, v16h b, v8f c) {
  return __builtin_amdgcn_wmma_f32_16x16x32_f16(
      /*neg_a=*/false, a, /*neg_b=*/false, b,
      /*c_mod=*/(short)0, c, /*reuse_a=*/false, /*reuse_b=*/false);
}

// A-fragment (16x32 f16, row m = lane%16):
//   lanes 0-15 : halves 0-15 = K 0-15
//   lanes16-31 : halves 0-15 = K 16-31
__device__ __forceinline__ v16h load_a_frag(const half_t* row, int hi) {
  v8h x0 = *(const v8h*)(row + (hi ? 16 : 0));
  v8h x1 = *(const v8h*)(row + 8 + (hi ? 16 : 0));
  v16h a;
#pragma unroll
  for (int i = 0; i < 8; ++i) { a[i] = x0[i]; a[8 + i] = x1[i]; }
  return a;
}

__device__ __forceinline__ v8f zero8() {
  v8f z = {0.f, 0.f, 0.f, 0.f, 0.f, 0.f, 0.f, 0.f};
  return z;
}

// ------------------------- fp32 -> fp16 convert ----------------------------
__global__ void __launch_bounds__(256)
ea_f2h(const float* __restrict__ s, half_t* __restrict__ d, int n) {
  int i = blockIdx.x * 256 + threadIdx.x;
  if (i < n) d[i] = (half_t)s[i];
}

// ------- WMMA GEMM, 64x64 tile/wave: C = A (MxK) * W^T (NxK), K=1024 -------
// 4 A-frags x 4 B-frags -> 16 wmma per k-step; each fragment reused 4x.
__global__ void __launch_bounds__(256)
ea_gemm_f16(const half_t* __restrict__ A, const half_t* __restrict__ W,
            half_t* __restrict__ C, int M, int N) {
  const int lane = threadIdx.x & 31;
  const int hi   = lane >> 4;
  const int ln   = lane & 15;
  const int wave = threadIdx.x >> 5;
  const int tiles_n = N >> 6;                      // 64-wide tiles
  const int tile = blockIdx.x * 8 + wave;
  if (tile >= (M >> 6) * tiles_n) return;
  const int tm = tile / tiles_n, tn = tile % tiles_n;

  const half_t* abase = A + (size_t)(tm * 64 + ln) * GEMM_K;
  const half_t* wbase = W + (size_t)(tn * 64 + ln) * GEMM_K + hi * 16;

  v8f acc[4][4];
#pragma unroll
  for (int i = 0; i < 4; ++i)
#pragma unroll
    for (int j = 0; j < 4; ++j) acc[i][j] = zero8();

  for (int kt = 0; kt < GEMM_K; kt += 32) {
    __builtin_prefetch(abase + kt + 256, 0, 0);    // global_prefetch_b8
    __builtin_prefetch(wbase + kt + 256, 0, 0);
    v16h af[4], bf[4];
#pragma unroll
    for (int i = 0; i < 4; ++i)                    // +i*16 rows: +32KB imm
      af[i] = load_a_frag(abase + (size_t)i * 16 * GEMM_K + kt, hi);
#pragma unroll
    for (int j = 0; j < 4; ++j)
      bf[j] = *(const v16h*)(wbase + (size_t)j * 16 * GEMM_K + kt);
#pragma unroll
    for (int i = 0; i < 4; ++i)
#pragma unroll
      for (int j = 0; j < 4; ++j)
        acc[i][j] = wmma16x16x32(af[i], bf[j], acc[i][j]);
  }

  // C/D layout: VGPR r -> row (r + hi*8), col = lane%16
#pragma unroll
  for (int i = 0; i < 4; ++i)
#pragma unroll
    for (int j = 0; j < 4; ++j) {
      half_t* crow = C + (size_t)(tm * 64 + i * 16 + hi * 8) * N
                       + tn * 64 + j * 16 + ln;
#pragma unroll
      for (int r = 0; r < 8; ++r) crow[(size_t)r * N] = (half_t)acc[i][j][r];
    }
}

// ---------------- final GEMM variant: f32 output + bias --------------------
__global__ void __launch_bounds__(256)
ea_gemm_f32b(const half_t* __restrict__ A, const half_t* __restrict__ W,
             float* __restrict__ C, const float* __restrict__ bias,
             int M, int N) {
  const int lane = threadIdx.x & 31;
  const int hi   = lane >> 4;
  const int ln   = lane & 15;
  const int wave = threadIdx.x >> 5;
  const int tiles_n = N >> 6;
  const int tile = blockIdx.x * 8 + wave;
  if (tile >= (M >> 6) * tiles_n) return;
  const int tm = tile / tiles_n, tn = tile % tiles_n;

  const half_t* abase = A + (size_t)(tm * 64 + ln) * GEMM_K;
  const half_t* wbase = W + (size_t)(tn * 64 + ln) * GEMM_K + hi * 16;

  v8f acc[4][4];
#pragma unroll
  for (int i = 0; i < 4; ++i)
#pragma unroll
    for (int j = 0; j < 4; ++j) acc[i][j] = zero8();

  for (int kt = 0; kt < GEMM_K; kt += 32) {
    __builtin_prefetch(abase + kt + 256, 0, 0);
    __builtin_prefetch(wbase + kt + 256, 0, 0);
    v16h af[4], bf[4];
#pragma unroll
    for (int i = 0; i < 4; ++i)
      af[i] = load_a_frag(abase + (size_t)i * 16 * GEMM_K + kt, hi);
#pragma unroll
    for (int j = 0; j < 4; ++j)
      bf[j] = *(const v16h*)(wbase + (size_t)j * 16 * GEMM_K + kt);
#pragma unroll
    for (int i = 0; i < 4; ++i)
#pragma unroll
      for (int j = 0; j < 4; ++j)
        acc[i][j] = wmma16x16x32(af[i], bf[j], acc[i][j]);
  }

#pragma unroll
  for (int i = 0; i < 4; ++i)
#pragma unroll
    for (int j = 0; j < 4; ++j) {
      const float bv = bias[tn * 64 + j * 16 + ln];
      float* crow = C + (size_t)(tm * 64 + i * 16 + hi * 8) * N
                      + tn * 64 + j * 16 + ln;
#pragma unroll
      for (int r = 0; r < 8; ++r) crow[(size_t)r * N] = acc[i][j][r] + bv;
    }
}

// ---------------- windowed attention: one block per (b, win, h) ------------
// Q: [NROWS][1024] f16 (head h at col h*64). KV: [NROWS][128] f16 (K|V).
// O: [NROWS][1024] f16.
__global__ void __launch_bounds__(256)
ea_attn(const half_t* __restrict__ Q, const half_t* __restrict__ KV,
        half_t* __restrict__ O) {
  __shared__ half_t Vt[HD][WIN + 8];       // V transposed: Vt[d][key]
  __shared__ half_t P[8][16][WIN + 8];     // per-wave probability row tile

  const int h   = blockIdx.x % NHEADS;
  const int win = (blockIdx.x / NHEADS) % NWIN;
  const int b   = blockIdx.x / (NHEADS * NWIN);
  const int l0  = b * SEQ + win * WIN;

  // Stage V transposed so PV B-fragments are contiguous in k.
  {
    const int key = threadIdx.x;                       // 0..255
    const half_t* vrow = KV + (size_t)(l0 + key) * 128 + HD;
#pragma unroll 4
    for (int d = 0; d < HD; ++d) Vt[d][key] = vrow[d];
  }
  __syncthreads();

  const int lane = threadIdx.x & 31;
  const int hi   = lane >> 4;
  const int ln   = lane & 15;
  const int wave = threadIdx.x >> 5;

  for (int pass = 0; pass < 2; ++pass) {
    const int rt = wave + pass * 8;                    // query row tile 0..15
    const half_t* qrow = Q + (size_t)(l0 + rt * 16 + ln) * DIMD + h * HD;

    // ---- S = Q Kt : 16 key tiles of 16, K-dim 64 -> 2 wmma each ----
    v8f acc[16];
#pragma unroll
    for (int t = 0; t < 16; ++t) acc[t] = zero8();
#pragma unroll
    for (int kk = 0; kk < HD; kk += 32) {
      v16h a = load_a_frag(qrow + kk, hi);             // reused 16x
#pragma unroll
      for (int t = 0; t < 16; ++t) {
        // B-frag b[j] = K[key = tile*16 + lane%16][kk + j + hi*16]
        const half_t* krow = KV + (size_t)(l0 + t * 16 + ln) * 128 + kk + hi * 16;
        v16h bf = *(const v16h*)krow;
        acc[t] = wmma16x16x32(a, bf, acc[t]);
      }
    }

    // ---- row softmax. C-layout: lane holds rows (i + hi*8), col lane%16;
    //      a full row spans one 16-lane group -> shfl_xor reduce over 1,2,4,8.
    float inv[8];
#pragma unroll
    for (int i = 0; i < 8; ++i) {
      float mx = -3.0e38f;
#pragma unroll
      for (int t = 0; t < 16; ++t) mx = fmaxf(mx, acc[t][i]);
      for (int m = 1; m < 16; m <<= 1) mx = fmaxf(mx, __shfl_xor(mx, m, 32));
      const float mxs = mx * QKSCALE;
      float sum = 0.f;
#pragma unroll
      for (int t = 0; t < 16; ++t) {
        float e = __expf(acc[t][i] * QKSCALE - mxs);
        acc[t][i] = e;
        sum += e;
      }
      for (int m = 1; m < 16; m <<= 1) sum += __shfl_xor(sum, m, 32);
      inv[i] = 1.0f / sum;
    }

    // ---- P -> LDS (row-major, A-fragment friendly; wave-private slab) ----
#pragma unroll
    for (int t = 0; t < 16; ++t)
#pragma unroll
      for (int i = 0; i < 8; ++i)
        P[wave][i + hi * 8][t * 16 + ln] = (half_t)(acc[t][i] * inv[i]);

    // ---- O = P V : 4 output tiles of 16 cols, K-dim 256 -> 8 wmma each ----
#pragma unroll
    for (int tn = 0; tn < 4; ++tn) {
      v8f oc = zero8();
#pragma unroll
      for (int kt = 0; kt < 8; ++kt) {
        v16h a  = load_a_frag(&P[wave][ln][kt * 32], hi);
        // b[j] = V[key = kt*32 + j + hi*16][d = tn*16 + lane%16] = Vt[d][key]
        v16h bf = *(const v16h*)&Vt[tn * 16 + ln][kt * 32 + hi * 16];
        oc = wmma16x16x32(a, bf, oc);
      }
      half_t* orow = O + (size_t)(l0 + rt * 16 + hi * 8) * DIMD + h * HD + tn * 16 + ln;
#pragma unroll
      for (int i = 0; i < 8; ++i) orow[(size_t)i * DIMD] = (half_t)oc[i];
    }
  }
}

// ---------------------------------------------------------------------------
extern "C" void kernel_launch(void* const* d_in, const int* in_sizes, int n_in,
                              void* d_out, int out_size, void* d_ws, size_t ws_size,
                              hipStream_t stream) {
  const float* x   = (const float*)d_in[0];
  const float* qw  = (const float*)d_in[1];
  const float* kvw = (const float*)d_in[2];
  const float* pw  = (const float*)d_in[3];
  const float* pb  = (const float*)d_in[4];
  float* out = (float*)d_out;

  // Workspace partition (~104 MB total)
  char* ws = (char*)d_ws;
  half_t* xh   = (half_t*)ws; ws += (size_t)NROWS * DIMD * 2;   // 32 MB
  half_t* qwh  = (half_t*)ws; ws += (size_t)DIMD * DIMD * 2;    //  2 MB
  half_t* kvwh = (half_t*)ws; ws += (size_t)128  * DIMD * 2;    // .25MB
  half_t* pwh  = (half_t*)ws; ws += (size_t)DIMD * DIMD * 2;    //  2 MB
  half_t* qh   = (half_t*)ws; ws += (size_t)NROWS * DIMD * 2;   // 32 MB
  half_t* kvh  = (half_t*)ws; ws += (size_t)NROWS * 128  * 2;   //  4 MB
  half_t* ah   = (half_t*)ws; ws += (size_t)NROWS * DIMD * 2;   // 32 MB

  // fp32 -> fp16 conversions
  {
    int n;
    n = NROWS * DIMD; ea_f2h<<<(n + 255) / 256, 256, 0, stream>>>(x,   xh,   n);
    n = DIMD * DIMD;  ea_f2h<<<(n + 255) / 256, 256, 0, stream>>>(qw,  qwh,  n);
    n = 128 * DIMD;   ea_f2h<<<(n + 255) / 256, 256, 0, stream>>>(kvw, kvwh, n);
    n = DIMD * DIMD;  ea_f2h<<<(n + 255) / 256, 256, 0, stream>>>(pw,  pwh,  n);
  }

  // Q = xh @ q_w^T   (16384 x 1024): 256x16 = 4096 wave tiles -> 512 blocks
  {
    int tiles = (NROWS / 64) * (DIMD / 64);
    ea_gemm_f16<<<(tiles + 7) / 8, 256, 0, stream>>>(xh, qwh, qh, NROWS, DIMD);
  }
  // KV = xh @ kv_w^T (16384 x 128): 256x2 = 512 wave tiles -> 64 blocks
  {
    int tiles = (NROWS / 64) * (128 / 64);
    ea_gemm_f16<<<(tiles + 7) / 8, 256, 0, stream>>>(xh, kvwh, kvh, NROWS, 128);
  }
  // Windowed attention: 4 * 16 * 16 = 1024 blocks
  ea_attn<<<BATCH * NWIN * NHEADS, 256, 0, stream>>>(qh, kvh, ah);

  // out = ah @ proj_w^T + proj_b  (f32 output)
  {
    int tiles = (NROWS / 64) * (DIMD / 64);
    ea_gemm_f32b<<<(tiles + 7) / 8, 256, 0, stream>>>(ah, pwh, out, pb, NROWS, DIMD);
  }
}
